// AdaptivePatchEmbedding_21981642621385
// MI455X (gfx1250) — compile-verified, matrix operand order checked by reading, not run on hardware
//
#include <hip/hip_runtime.h>

typedef __attribute__((ext_vector_type(16))) __bf16 v16bf;
typedef __attribute__((ext_vector_type(8)))  float  v8f;

#define IMG_H 1024
#define IMG_W 1024
#define D_DIM 768
#define NTILES 48        // 768 / 16
#define NT_PER_WAVE 6    // 48 ntiles / 8 waves
#define KC 24            // K tiles for K=768 (24 * 32)

// ---------------------------------------------------------------------------
// Fragment index helpers (wave32 WMMA 16x16x32 bf16).
//   A/B fragment: 32 lanes x 16 halves contiguous per (ktile, tile).
//   lane = ((kin&15)>=8)*16 + (row|col & 15); e = (kin&7) + 8*(kin>=16)
// ---------------------------------------------------------------------------
__device__ __forceinline__ void store_apack(__bf16* __restrict__ outP,
                                            int grow, int n, int ggShift,
                                            int outKT, float v)
{
    // producer row 'grow' of a [R,768] result viewed as consumer A[arow, kk]
    const int arow = grow >> ggShift;
    const int hw   = grow & ((1 << ggShift) - 1);
    const int kk   = hw * 768 + n;
    const int mt   = arow >> 4, mm = arow & 15;
    const int kt   = kk >> 5,   kin = kk & 31;
    const int lane = (((kin & 15) >> 3) << 4) + mm;
    const int e    = (kin & 7) + ((kin >> 4) << 3);
    outP[(((size_t)mt * outKT + kt) * 32 + lane) * 16 + e] = (__bf16)v;
}

// ---------------------------------------------------------------------------
// Weight pre-pack into B-fragment layout (bf16).
// mode 0: src is [N=768][Ktot] row-major (conv-as-matmul / mlp_w)
// mode 1: src is agg_w [O=768][768][g][g]; k = (h*g+w)*768 + d
// ---------------------------------------------------------------------------
__global__ __launch_bounds__(256)
void pack_w(const float* __restrict__ src, __bf16* __restrict__ dst,
            int Ktot, int mode, int gg)
{
    const long long tid = (long long)blockIdx.x * 256 + threadIdx.x;
    const long long total = (long long)Ktot * 768;
    if (tid >= total) return;
    const int e     = (int)(tid & 15);
    const int lane  = (int)((tid >> 4) & 31);
    const int ntile = (int)((tid >> 9) % 48);
    const int ktg   = (int)(tid / (16 * 32 * 48));
    const int n     = ntile * 16 + (lane & 15);
    const int kin   = (e & 7) + ((e >> 3) << 4) + ((lane >> 4) << 3);
    const int k     = ktg * 32 + kin;
    size_t idx;
    if (mode == 0) {
        idx = (size_t)n * Ktot + k;
    } else {
        const int kd = k / 768;              // h*g + w
        const int km = k - kd * 768;         // d
        idx = ((size_t)n * 768 + km) * gg + kd;
    }
    dst[tid] = (__bf16)src[idx];
}

// ---------------------------------------------------------------------------
// Gather GEMM (K = 768 fixed): 16 rows x 768 cols per block, A via LDS.
// MODE 0: 16x16x3 patch at coords[row]
// MODE 1: g x g sub-patch gather (row = n*g*g + h*g + w)
// MODE 2: 2x2-avg-pooled 32x32 patch   (exact bilinear 32->16)
// MODE 3: pooled 64x64, pixels (4y+1,4y+2)x(4x+1,4x+2) (64->16)
// OUTMODE 0: f32 row-major to outF (C init = bias + optional zsrc residual)
// OUTMODE 1: bf16 A-fragment-packed to outP (for the next GEMM stage)
// Inner loop is software-pipelined: next A/B fragments are fetched before the
// WMMA that consumes the current ones (waits relax from drain to depth-1).
// ---------------------------------------------------------------------------
template <int MODE, int OUTMODE>
__global__ __launch_bounds__(256)
void gemm_gather(const float* __restrict__ image,
                 const int*   __restrict__ coords,
                 const __bf16* __restrict__ Bp,
                 const float* __restrict__ bias,
                 const float* __restrict__ zsrc,
                 float*  __restrict__ outF,
                 __bf16* __restrict__ outP, int ggShift, int outKT,
                 int g)
{
    __shared__ __attribute__((aligned(32))) __bf16 Alds[KC * 32 * 16];

    const int tid  = threadIdx.x;
    const int lane = tid & 31;
    const int wave = tid >> 5;
    const int row0 = (int)blockIdx.x * 16;
    const int ncol = lane & 15;
    const int mhi  = (lane >> 4) << 3;

    v8f acc[NT_PER_WAVE];
#pragma unroll
    for (int t = 0; t < NT_PER_WAVE; ++t) {
        const int n = (wave * NT_PER_WAVE + t) * 16 + ncol;
        const float bv = bias[n];
#pragma unroll
        for (int r = 0; r < 8; ++r) {
            float z = 0.f;
            if (OUTMODE == 0 && zsrc)
                z = zsrc[(size_t)(row0 + r + mhi) * D_DIM + n];
            acc[t][r] = bv + z;
        }
    }

    // ---- stage A (16 rows x 768 K) into LDS in A-fragment layout
#pragma unroll 1
    for (int mm = 0; mm < 16; ++mm) {
        const int row = row0 + mm;
        int cy, cx;
        if (MODE == 1) {
            const int gg = g * g;
            const int n0 = row / gg;
            const int rr = row - n0 * gg;
            const int h = rr / g, w = rr - h * g;
            cy = coords[2 * n0]     + h * 16;
            cx = coords[2 * n0 + 1] + w * 16;
        } else {
            cy = coords[2 * row];
            cx = coords[2 * row + 1];
        }
        for (int kl = tid; kl < 768; kl += 256) {
            const int c   = kl >> 8;          // k = c*256 + py*16 + px
            const int rem = kl & 255;
            const int py  = rem >> 4;
            const int px  = rem & 15;
            float v;
            if (MODE == 0 || MODE == 1) {
                v = image[(size_t)c * (IMG_H * IMG_W) +
                          (size_t)(cy + py) * IMG_W + (cx + px)];
            } else if (MODE == 2) {
                const float* p = image + (size_t)c * (IMG_H * IMG_W) +
                                 (size_t)(cy + 2 * py) * IMG_W + (cx + 2 * px);
                v = 0.25f * (p[0] + p[1] + p[IMG_W] + p[IMG_W + 1]);
            } else { // MODE 3
                const float* p = image + (size_t)c * (IMG_H * IMG_W) +
                                 (size_t)(cy + 4 * py + 1) * IMG_W + (cx + 4 * px + 1);
                v = 0.25f * (p[0] + p[1] + p[IMG_W] + p[IMG_W + 1]);
            }
            const int ktl = kl >> 5;
            const int kin = kl & 31;
            const int sl  = (((kin & 15) >> 3) << 4) + mm;
            const int e   = (kin & 7) + ((kin >> 4) << 3);
            Alds[(ktl * 32 + sl) * 16 + e] = (__bf16)v;
        }
    }
    __syncthreads();

    // ---- fully unrolled, software-pipelined K loop
    const __bf16* __restrict__ bb =
        Bp + ((size_t)wave * NT_PER_WAVE * 32 + lane) * 16;
    const __bf16* __restrict__ ab = &Alds[lane * 16];

    v16bf a = *(const v16bf*)ab;
    v16bf b = *(const v16bf*)bb;
#pragma unroll
    for (int kt = 0; kt < KC; ++kt) {
        v16bf a_next = a;
        if (kt + 1 < KC)
            a_next = *(const v16bf*)(ab + (kt + 1) * (32 * 16));
#pragma unroll
        for (int t = 0; t < NT_PER_WAVE; ++t) {
            v16bf b_next = b;
            if (t + 1 < NT_PER_WAVE)
                b_next = *(const v16bf*)(bb + kt * (NTILES * 512) + (t + 1) * 512);
            else if (kt + 1 < KC)
                b_next = *(const v16bf*)(bb + (kt + 1) * (NTILES * 512));
            acc[t] = __builtin_amdgcn_wmma_f32_16x16x32_bf16(
                false, a, false, b, (short)0, acc[t], false, false);
            b = b_next;
        }
        a = a_next;
    }

    // ---- epilogue
#pragma unroll
    for (int t = 0; t < NT_PER_WAVE; ++t) {
        const int n = (wave * NT_PER_WAVE + t) * 16 + ncol;
#pragma unroll
        for (int r = 0; r < 8; ++r) {
            const int grow = row0 + r + mhi;
            if (OUTMODE == 0) outF[(size_t)grow * D_DIM + n] = acc[t][r];
            else              store_apack(outP, grow, n, ggShift, outKT, acc[t][r]);
        }
    }
}

// ---------------------------------------------------------------------------
// Packed-A GEMM: A already in A-fragment layout in global memory.
// No LDS, no barriers: each lane streams its contiguous 32B fragment.
// Software-pipelined depth-1 (over-reads one fragment; ws tail is padded).
// OUTMODE 0: f32 row-major (bias only). OUTMODE 1: bf16 A-packed (ggShift=0).
// ---------------------------------------------------------------------------
template <int OUTMODE>
__global__ __launch_bounds__(256)
void gemm_packed(const __bf16* __restrict__ Ap, int KT,
                 const __bf16* __restrict__ Bp,
                 const float* __restrict__ bias,
                 float*  __restrict__ outF,
                 __bf16* __restrict__ outP, int outKT)
{
    const int tid  = threadIdx.x;
    const int lane = tid & 31;
    const int wave = tid >> 5;
    const int row0 = (int)blockIdx.x * 16;
    const int ncol = lane & 15;
    const int mhi  = (lane >> 4) << 3;

    v8f acc[NT_PER_WAVE];
#pragma unroll
    for (int t = 0; t < NT_PER_WAVE; ++t) {
        const float bv = bias[(wave * NT_PER_WAVE + t) * 16 + ncol];
#pragma unroll
        for (int r = 0; r < 8; ++r) acc[t][r] = bv;
    }

    const __bf16* __restrict__ ap =
        Ap + (((size_t)blockIdx.x * KT) * 32 + lane) * 16;
    const __bf16* __restrict__ bp =
        Bp + ((size_t)wave * NT_PER_WAVE * 32 + lane) * 16;

    v16bf a = *(const v16bf*)ap;
    v16bf b = *(const v16bf*)bp;
#pragma unroll 2
    for (int kt = 0; kt < KT; ++kt) {
        const v16bf a_next = *(const v16bf*)(ap + 512);   // 1-frag over-read OK
#pragma unroll
        for (int t = 0; t < NT_PER_WAVE; ++t) {
            v16bf b_next;
            if (t + 1 < NT_PER_WAVE)
                b_next = *(const v16bf*)(bp + (t + 1) * 512);
            else
                b_next = *(const v16bf*)(bp + NTILES * 512); // over-read OK
            acc[t] = __builtin_amdgcn_wmma_f32_16x16x32_bf16(
                false, a, false, b, (short)0, acc[t], false, false);
            b = b_next;
        }
        a = a_next;
        ap += 512;
        bp += NTILES * 512;
    }

#pragma unroll
    for (int t = 0; t < NT_PER_WAVE; ++t) {
        const int n = (wave * NT_PER_WAVE + t) * 16 + ncol;
#pragma unroll
        for (int r = 0; r < 8; ++r) {
            const int grow = row0 + r + mhi;
            if (OUTMODE == 0) outF[(size_t)grow * D_DIM + n] = acc[t][r];
            else              store_apack(outP, grow, n, 0, outKT, acc[t][r]);
        }
    }
}

// ---------------------------------------------------------------------------
extern "C" void kernel_launch(void* const* d_in, const int* in_sizes, int n_in,
                              void* d_out, int out_size, void* d_ws, size_t ws_size,
                              hipStream_t stream)
{
    (void)in_sizes; (void)n_in; (void)out_size; (void)ws_size;
    const float* image   = (const float*)d_in[0];
    const int*   coords0 = (const int*)d_in[1];
    const int*   coords1 = (const int*)d_in[2];
    const int*   coords2 = (const int*)d_in[3];
    const float* base_w  = (const float*)d_in[4];
    const float* base_b  = (const float*)d_in[5];
    const float* agg_w1  = (const float*)d_in[6];
    const float* agg_b1  = (const float*)d_in[7];
    const float* agg_w2  = (const float*)d_in[8];
    const float* agg_b2  = (const float*)d_in[9];
    const float* mlp_w1  = (const float*)d_in[10];
    const float* mlp_b1  = (const float*)d_in[11];
    const float* mlp_w2  = (const float*)d_in[12];
    const float* mlp_b2  = (const float*)d_in[13];
    float* out = (float*)d_out;

    char* wsc = (char*)d_ws;
    size_t off = 0;
    auto take = [&](size_t bytes) -> void* {
        void* p = wsc + off;
        off = (off + bytes + 255) & ~(size_t)255;
        return p;
    };
    __bf16* Wb    = (__bf16*)take((size_t)768  * 768 * 2);
    __bf16* W1    = (__bf16*)take((size_t)3072 * 768 * 2);
    __bf16* W2    = (__bf16*)take((size_t)12288* 768 * 2);
    __bf16* M1    = (__bf16*)take((size_t)768  * 768 * 2);
    __bf16* M2    = (__bf16*)take((size_t)768  * 768 * 2);
    __bf16* sub1p = (__bf16*)take((size_t)16384* 768 * 2);  // A-packed [4096,3072]
    __bf16* sub2p = (__bf16*)take((size_t)32768* 768 * 2);  // A-packed [2048,12288]
    __bf16* agg1p = (__bf16*)take((size_t)4096 * 768 * 2);  // A-packed [4096,768]
    __bf16* agg2p = (__bf16*)take((size_t)2048 * 768 * 2);  // A-packed [2048,768]
    (void)take(2048);  // tail pad: pipelined kernels over-read one fragment

    // ---- pack weights into B-fragment layout
    pack_w<<<(768*768   + 255)/256, 256, 0, stream>>>(base_w, Wb, 768,   0, 1);
    pack_w<<<(3072*768  + 255)/256, 256, 0, stream>>>(agg_w1, W1, 3072,  1, 4);
    pack_w<<<(12288*768 + 255)/256, 256, 0, stream>>>(agg_w2, W2, 12288, 1, 16);
    pack_w<<<(768*768   + 255)/256, 256, 0, stream>>>(mlp_w1, M1, 768,   0, 1);
    pack_w<<<(768*768   + 255)/256, 256, 0, stream>>>(mlp_w2, M2, 768,   0, 1);

    float* outE1 = out + (size_t)8192  * 768;
    float* outE2 = out + (size_t)12288 * 768;

    // ---- e0: direct patch embed -> d_out rows [0, 8192)
    gemm_gather<0, 0><<<512, 256, 0, stream>>>(image, coords0, Wb, base_b,
                                               nullptr, out, nullptr, 0, 0, 0);
    // ---- sub-patch embeds, written A-packed for the agg GEMMs
    gemm_gather<1, 1><<<1024, 256, 0, stream>>>(image, coords1, Wb, base_b,
                                                nullptr, nullptr, sub1p, 2, 96, 2);
    gemm_gather<1, 1><<<2048, 256, 0, stream>>>(image, coords2, Wb, base_b,
                                                nullptr, nullptr, sub2p, 4, 384, 4);
    // ---- aggregator convs as big-K packed GEMMs -> A-packed for MLP
    gemm_packed<1><<<256, 256, 0, stream>>>(sub1p, 96,  W1, agg_b1,
                                            nullptr, agg1p, 24);
    gemm_packed<1><<<128, 256, 0, stream>>>(sub2p, 384, W2, agg_b2,
                                            nullptr, agg2p, 24);
    // ---- zero-init MLPs: z written straight into d_out (C-init for next pass)
    gemm_packed<0><<<256, 256, 0, stream>>>(agg1p, 24, M1, mlp_b1,
                                            outE1, nullptr, 0);
    gemm_packed<0><<<128, 256, 0, stream>>>(agg2p, 24, M2, mlp_b2,
                                            outE2, nullptr, 0);
    // ---- resized (exact bilinear = pooled gather) base embeds, + z residual
    gemm_gather<2, 0><<<256, 256, 0, stream>>>(image, coords1, Wb, base_b,
                                               outE1, outE1, nullptr, 0, 0, 0);
    gemm_gather<3, 0><<<128, 256, 0, stream>>>(image, coords2, Wb, base_b,
                                               outE2, outE2, nullptr, 0, 0, 0);
}
